// MultiHeadLatentAttention_66554813219260
// MI455X (gfx1250) — compile-verified, matrix operand order checked by reading, not run on hardware
//
#include <hip/hip_runtime.h>
#include <hip/hip_bf16.h>

typedef __bf16 bf16_t;
typedef __attribute__((ext_vector_type(16))) __bf16       v16bf;
typedef __attribute__((ext_vector_type(8)))  float        v8f;
typedef __attribute__((ext_vector_type(4)))  unsigned int u32x4;

#define NH_   16
#define NOPE_ 128
#define ROPE_ 64
#define VD_   128
#define DTOT_ 192   // NOPE + ROPE
#define HID_  2048
#define QC_   1536
#define KVC_  512
#define B_    2
#define S_    2048
#define BS_   (B_ * S_)

union FragU { v16bf v; u32x4 q[2]; };

__device__ __forceinline__ v16bf load_frag16(const bf16_t* p) {
  FragU u;
  u.q[0] = *(const u32x4*)(p);        // 8 bf16: K = koff .. koff+7
  u.q[1] = *(const u32x4*)(p + 16);   // 8 bf16: K = koff+16 .. koff+23
  return u.v;
}

// ---------------------------------------------------------------------------
// Elementwise f32 -> bf16 conversion (grid-stride)
// ---------------------------------------------------------------------------
__global__ void cvt_f32_bf16(const float* __restrict__ in, bf16_t* __restrict__ out, size_t n) {
  for (size_t i = (size_t)blockIdx.x * blockDim.x + threadIdx.x; i < n;
       i += (size_t)gridDim.x * blockDim.x)
    out[i] = (bf16_t)in[i];
}

// ---------------------------------------------------------------------------
// Row-wise RMSNorm: in (rows x C) f32 -> out bf16, with gain w
// ---------------------------------------------------------------------------
__global__ void rmsnorm_rows(const float* __restrict__ in, const float* __restrict__ w,
                             bf16_t* __restrict__ out, int C) {
  __shared__ float red[256];
  const size_t row = blockIdx.x;
  const float* r = in + row * (size_t)C;
  float s = 0.f;
  for (int c = threadIdx.x; c < C; c += 256) { float v = r[c]; s += v * v; }
  red[threadIdx.x] = s;
  __syncthreads();
  for (int off = 128; off > 0; off >>= 1) {
    if ((int)threadIdx.x < off) red[threadIdx.x] += red[threadIdx.x + off];
    __syncthreads();
  }
  const float inv = rsqrtf(red[0] / (float)C + 1e-6f);
  for (int c = threadIdx.x; c < C; c += 256)
    out[row * (size_t)C + c] = (bf16_t)(r[c] * inv * w[c]);
}

// ---------------------------------------------------------------------------
// WMMA GEMM: C[m,n] = sum_k A[m,k] * W[n,k]
//   A: (M x K) bf16 row-major, W: (N x K) bf16 row-major, C: (M x N) f32
//   one wave -> 16(M) x 64(N) tile; 256 thr = 8 waves / block
// ---------------------------------------------------------------------------
__global__ void gemm_bf16_wmma(const bf16_t* __restrict__ A, const bf16_t* __restrict__ W,
                               float* __restrict__ C, int M, int N, int K) {
  const int lane = threadIdx.x & 31;
  const int wid  = blockIdx.x * (blockDim.x >> 5) + (threadIdx.x >> 5);
  const int ngrp = N >> 6;
  const int tm   = wid / ngrp;
  const int tn   = wid - tm * ngrp;
  if (tm >= (M >> 4)) return;

  const int nloc = lane & 15;
  const int koff = (lane >> 4) << 3;

  const bf16_t* arow = A + (size_t)(tm * 16 + nloc) * K + koff;
  const bf16_t* wrow = W + (size_t)(tn * 64 + nloc) * K + koff;

  v8f acc[4];
#pragma unroll
  for (int t = 0; t < 4; t++)
#pragma unroll
    for (int r = 0; r < 8; r++) acc[t][r] = 0.f;

  for (int k0 = 0; k0 < K; k0 += 32) {
    if (k0 + 256 < K) {          // WGP-scope prefetch ahead of demand loads
      __builtin_prefetch(arow + k0 + 256, 0, 3);
      __builtin_prefetch(wrow + k0 + 256, 0, 3);
    }
    v16bf a = load_frag16(arow + k0);
#pragma unroll
    for (int t = 0; t < 4; t++) {
      v16bf b = load_frag16(wrow + (size_t)t * 16 * K + k0);
      acc[t] = __builtin_amdgcn_wmma_f32_16x16x32_bf16(false, a, false, b, (short)0,
                                                       acc[t], false, false);
    }
  }

  const int mh = (lane >> 4) * 8;
#pragma unroll
  for (int t = 0; t < 4; t++) {
    const int n = tn * 64 + t * 16 + nloc;
#pragma unroll
    for (int r = 0; r < 8; r++)
      C[(size_t)(tm * 16 + mh + r) * N + n] = acc[t][r];
  }
}

// ---------------------------------------------------------------------------
// Build Q (B,NH,S,192) bf16: nope part copied, rope part rotated per-head
// ---------------------------------------------------------------------------
__global__ void build_q(const float* __restrict__ qnope, const float* __restrict__ qrope,
                        bf16_t* __restrict__ qcat) {
  const int bs = blockIdx.x;
  const int s  = bs % S_;
  const int b  = bs / S_;
  for (int idx = threadIdx.x; idx < NH_ * DTOT_; idx += blockDim.x) {
    const int h = idx / DTOT_, d = idx % DTOT_;
    float val;
    if (d < NOPE_) {
      val = qnope[(size_t)bs * (NH_ * NOPE_) + h * NOPE_ + d];
    } else {
      const int i  = d - NOPE_;
      const int fi = i & 31;
      const float fr = (float)s * __powf(10000.f, -(float)(2 * fi) / (float)ROPE_);
      const float cs = __cosf(fr), sn = __sinf(fr);
      const float x0 = qrope[(size_t)bs * (NH_ * ROPE_) + h * ROPE_ + i];
      const float xr = (i < 32)
          ? -qrope[(size_t)bs * (NH_ * ROPE_) + h * ROPE_ + i + 32]
          :  qrope[(size_t)bs * (NH_ * ROPE_) + h * ROPE_ + i - 32];
      val = x0 * cs + xr * sn;
    }
    qcat[(((size_t)(b * NH_ + h)) * S_ + s) * DTOT_ + d] = (bf16_t)val;
  }
}

// ---------------------------------------------------------------------------
// Build K (B,NH,S,192) bf16: rope part shared across heads
// ---------------------------------------------------------------------------
__global__ void build_k(const float* __restrict__ knope, const float* __restrict__ krope,
                        bf16_t* __restrict__ kcat) {
  __shared__ float rot[ROPE_];
  const int bs = blockIdx.x;
  const int s  = bs % S_;
  const int b  = bs / S_;
  if ((int)threadIdx.x < ROPE_) {
    const int i  = threadIdx.x;
    const int fi = i & 31;
    const float fr = (float)s * __powf(10000.f, -(float)(2 * fi) / (float)ROPE_);
    const float x0 = krope[(size_t)bs * ROPE_ + i];
    const float xr = (i < 32) ? -krope[(size_t)bs * ROPE_ + i + 32]
                              :  krope[(size_t)bs * ROPE_ + i - 32];
    rot[i] = x0 * __cosf(fr) + xr * __sinf(fr);
  }
  __syncthreads();
  for (int idx = threadIdx.x; idx < NH_ * DTOT_; idx += blockDim.x) {
    const int h = idx / DTOT_, d = idx % DTOT_;
    const float val = (d < NOPE_)
        ? knope[(size_t)bs * (NH_ * NOPE_) + h * NOPE_ + d]
        : rot[d - NOPE_];
    kcat[(((size_t)(b * NH_ + h)) * S_ + s) * DTOT_ + d] = (bf16_t)val;
  }
}

// ---------------------------------------------------------------------------
// V (B,S,NH*VD) f32  ->  Vt (B,NH,VD,S) bf16  (S-contiguous for PV B-frags)
// ---------------------------------------------------------------------------
__global__ void build_vt(const float* __restrict__ v, bf16_t* __restrict__ vt) {
  const size_t total = (size_t)B_ * NH_ * VD_ * S_;
  for (size_t idx = (size_t)blockIdx.x * blockDim.x + threadIdx.x; idx < total;
       idx += (size_t)gridDim.x * blockDim.x) {
    const int    sidx = (int)(idx % S_);
    const size_t r    = idx / S_;
    const int    vd   = (int)(r % VD_);
    const size_t r2   = r / VD_;
    const int    h    = (int)(r2 % NH_);
    const int    b    = (int)(r2 / NH_);
    vt[idx] = (bf16_t)v[((size_t)(b * S_ + sidx)) * (NH_ * VD_) + h * VD_ + vd];
  }
}

// ---------------------------------------------------------------------------
// Flash attention with async-to-LDS staging of shared K/V tiles.
// grid = B*NH*(S/128), block = 256 (8 waves); each wave owns a 16-row Q tile.
// Per 32-key step: DMA next K (32x192) + V (128x32) tile into the other LDS
// buffer (ASYNCcnt), 12 QK WMMAs + online softmax + 8 PV WMMAs from LDS.
// ---------------------------------------------------------------------------
#define KT_ELE (32 * DTOT_)   // 6144 bf16 = 12 KB
#define VT_ELE (VD_ * 32)     // 4096 bf16 =  8 KB
#define PT_ELE (8 * 16 * 32)  // 4096 bf16 =  8 KB

__global__ void mla_attn(const bf16_t* __restrict__ q, const bf16_t* __restrict__ k,
                         const bf16_t* __restrict__ vt, bf16_t* __restrict__ attn) {
  __shared__ bf16_t smem[2 * KT_ELE + 2 * VT_ELE + PT_ELE];
  bf16_t* kbuf0 = smem;
  bf16_t* kbuf1 = smem + KT_ELE;
  bf16_t* vbuf0 = smem + 2 * KT_ELE;
  bf16_t* vbuf1 = smem + 2 * KT_ELE + VT_ELE;
  bf16_t* pls   = smem + 2 * KT_ELE + 2 * VT_ELE;

  const int lane = threadIdx.x & 31;
  const int w    = threadIdx.x >> 5;
  const int tiles_per_bh = S_ / 128;
  const int bh = blockIdx.x / tiles_per_bh;
  const int qt = blockIdx.x % tiles_per_bh;
  const int qbase = qt * 128 + w * 16;

  const bf16_t* qp = q  + (size_t)bh * S_ * DTOT_;
  const bf16_t* kp = k  + (size_t)bh * S_ * DTOT_;
  const bf16_t* vp = vt + (size_t)bh * VD_ * S_;

  const int nloc = lane & 15;
  const int koff = (lane >> 4) << 3;
  const float scale = 0.07216878364870323f;  // (NOPE+ROPE)^-0.5

  // LDS byte address of a pointer into smem (static LDS block starts at 0)
  auto ldsaddr = [&](const bf16_t* p) -> unsigned {
    return (unsigned)((const char*)p - (const char*)smem);
  };
  // cooperative async DMA of one 32-key K/V tile into LDS (ASYNCcnt tracked)
  auto stage = [&](int kb, bf16_t* kdst, bf16_t* vdst) {
    for (int c = threadIdx.x; c < 768; c += 256) {        // K: 32 x 192 in 16B chunks
      const int row = c / 24, col = (c % 24) * 8;
      asm volatile("global_load_async_to_lds_b128 %0, %1, off"
                   :: "v"(ldsaddr(kdst + row * DTOT_ + col)),
                      "v"(kp + (size_t)(kb + row) * DTOT_ + col)
                   : "memory");
    }
    for (int c = threadIdx.x; c < 512; c += 256) {        // V: 128 x 32 in 16B chunks
      const int row = c >> 2, col = (c & 3) * 8;
      asm volatile("global_load_async_to_lds_b128 %0, %1, off"
                   :: "v"(ldsaddr(vdst + row * 32 + col)),
                      "v"(vp + (size_t)row * S_ + kb + col)
                   : "memory");
    }
  };

  v16bf qf[6];
#pragma unroll
  for (int i = 0; i < 6; i++)
    qf[i] = load_frag16(qp + (size_t)(qbase + nloc) * DTOT_ + i * 32 + koff);

  float m[8], l[8];
  v8f acc[8];
#pragma unroll
  for (int r = 0; r < 8; r++) { m[r] = -1e30f; l[r] = 0.f; }
#pragma unroll
  for (int j = 0; j < 8; j++)
#pragma unroll
    for (int r = 0; r < 8; r++) acc[j][r] = 0.f;

  bf16_t* myp = pls + w * 16 * 32;

  stage(0, kbuf0, vbuf0);
  asm volatile("s_wait_asynccnt 0x0" ::: "memory");
  __syncthreads();

  int cur = 0;
  for (int kb = 0; kb < S_; kb += 32) {
    const bf16_t* kc = cur ? kbuf1 : kbuf0;
    const bf16_t* vc = cur ? vbuf1 : vbuf0;
    if (kb + 32 < S_)
      stage(kb + 32, cur ? kbuf0 : kbuf1, cur ? vbuf0 : vbuf1);

    v8f c0, c1;
#pragma unroll
    for (int r = 0; r < 8; r++) { c0[r] = 0.f; c1[r] = 0.f; }

#pragma unroll
    for (int i = 0; i < 6; i++) {
      v16bf k0f = load_frag16(kc + nloc * DTOT_ + i * 32 + koff);
      c0 = __builtin_amdgcn_wmma_f32_16x16x32_bf16(false, qf[i], false, k0f, (short)0,
                                                   c0, false, false);
      v16bf k1f = load_frag16(kc + (16 + nloc) * DTOT_ + i * 32 + koff);
      c1 = __builtin_amdgcn_wmma_f32_16x16x32_bf16(false, qf[i], false, k1f, (short)0,
                                                   c1, false, false);
    }

    float alpha[8];
#pragma unroll
    for (int r = 0; r < 8; r++) {
      const float s0 = c0[r] * scale, s1 = c1[r] * scale;
      float mx = fmaxf(s0, s1);
      for (int d = 1; d < 16; d <<= 1) mx = fmaxf(mx, __shfl_xor(mx, d, 32));
      const float mn = fmaxf(m[r], mx);
      alpha[r] = __expf(m[r] - mn);
      m[r] = mn;
      const float e0 = __expf(s0 - mn), e1 = __expf(s1 - mn);
      float rs = e0 + e1;
      for (int d = 1; d < 16; d <<= 1) rs += __shfl_xor(rs, d, 32);
      l[r] = l[r] * alpha[r] + rs;
      c0[r] = e0; c1[r] = e1;
    }
#pragma unroll
    for (int j = 0; j < 8; j++)
#pragma unroll
      for (int r = 0; r < 8; r++) acc[j][r] *= alpha[r];

    // P tile: C-layout -> A-layout transpose through LDS
#pragma unroll
    for (int r = 0; r < 8; r++) {
      const int row = r + ((lane >> 4) << 3);
      myp[row * 32 + nloc]      = (bf16_t)c0[r];
      myp[row * 32 + 16 + nloc] = (bf16_t)c1[r];
    }
    __syncthreads();

    v16bf pf = load_frag16(myp + nloc * 32 + koff);
#pragma unroll
    for (int j = 0; j < 8; j++) {
      v16bf vf = load_frag16(vc + (j * 16 + nloc) * 32 + koff);
      acc[j] = __builtin_amdgcn_wmma_f32_16x16x32_bf16(false, pf, false, vf, (short)0,
                                                       acc[j], false, false);
    }

    asm volatile("s_wait_asynccnt 0x0" ::: "memory");  // next tile DMA done (this wave)
    __syncthreads();                                   // published to all waves
    cur ^= 1;
  }

  const int b = bh >> 4, h = bh & 15;
  const int mh = (lane >> 4) * 8;
#pragma unroll
  for (int j = 0; j < 8; j++)
#pragma unroll
    for (int r = 0; r < 8; r++) {
      const int srow = qbase + mh + r;
      const int col  = h * VD_ + j * 16 + nloc;
      attn[((size_t)(b * S_ + srow)) * (NH_ * VD_) + col] = (bf16_t)(acc[j][r] / l[r]);
    }
}

// ---------------------------------------------------------------------------
extern "C" void kernel_launch(void* const* d_in, const int* in_sizes, int n_in,
                              void* d_out, int out_size, void* d_ws, size_t ws_size,
                              hipStream_t stream) {
  const float* x       = (const float*)d_in[0];
  const float* wqdown  = (const float*)d_in[1];
  const float* qnormw  = (const float*)d_in[2];
  const float* wqnope  = (const float*)d_in[3];
  const float* wqrope  = (const float*)d_in[4];
  const float* wkvdown = (const float*)d_in[5];
  const float* kvnormw = (const float*)d_in[6];
  const float* wknope  = (const float*)d_in[7];
  const float* wvup    = (const float*)d_in[8];
  const float* wkrope  = (const float*)d_in[9];
  const float* wo      = (const float*)d_in[10];
  float* out = (float*)d_out;

  char* base = (char*)d_ws;
  size_t off = 0;
  auto alloc = [&](size_t bytes) -> char* {
    char* p = base + off;
    off += (bytes + 255) & ~(size_t)255;
    return p;
  };

  bf16_t* x_bf    = (bf16_t*)alloc((size_t)BS_ * HID_ * 2);
  bf16_t* wqd_bf  = (bf16_t*)alloc((size_t)QC_ * HID_ * 2);
  bf16_t* wqn_bf  = (bf16_t*)alloc((size_t)NH_ * NOPE_ * QC_ * 2);
  bf16_t* wqr_bf  = (bf16_t*)alloc((size_t)NH_ * ROPE_ * QC_ * 2);
  bf16_t* wkvd_bf = (bf16_t*)alloc((size_t)KVC_ * HID_ * 2);
  bf16_t* wkn_bf  = (bf16_t*)alloc((size_t)NH_ * NOPE_ * KVC_ * 2);
  bf16_t* wvu_bf  = (bf16_t*)alloc((size_t)NH_ * VD_ * KVC_ * 2);
  bf16_t* wkr_bf  = (bf16_t*)alloc((size_t)ROPE_ * HID_ * 2);
  bf16_t* wo_bf   = (bf16_t*)alloc((size_t)HID_ * NH_ * VD_ * 2);
  float*  cq_f    = (float*)alloc((size_t)BS_ * QC_ * 4);
  bf16_t* cq_bf   = (bf16_t*)alloc((size_t)BS_ * QC_ * 2);
  float*  ckv_f   = (float*)alloc((size_t)BS_ * KVC_ * 4);
  bf16_t* ckv_bf  = (bf16_t*)alloc((size_t)BS_ * KVC_ * 2);
  float*  krope_f = (float*)alloc((size_t)BS_ * ROPE_ * 4);
  float*  tmpA    = (float*)alloc((size_t)BS_ * 2048 * 4);  // qnope / knope / v
  float*  tmpB    = (float*)alloc((size_t)BS_ * 1024 * 4);  // qrope
  bf16_t* qcat    = (bf16_t*)alloc((size_t)B_ * NH_ * S_ * DTOT_ * 2);
  bf16_t* kcat    = (bf16_t*)alloc((size_t)B_ * NH_ * S_ * DTOT_ * 2);
  bf16_t* vt      = (bf16_t*)alloc((size_t)B_ * NH_ * VD_ * S_ * 2);
  bf16_t* attn_bf = (bf16_t*)alloc((size_t)BS_ * NH_ * VD_ * 2);

  auto cvt = [&](const float* src, bf16_t* dst, size_t n) {
    cvt_f32_bf16<<<1024, 256, 0, stream>>>(src, dst, n);
  };
  auto gemm = [&](const bf16_t* A, const bf16_t* W, float* C, int M, int N, int K) {
    int waves  = (M / 16) * (N / 64);
    int blocks = (waves + 7) / 8;
    gemm_bf16_wmma<<<blocks, 256, 0, stream>>>(A, W, C, M, N, K);
  };

  // 0) precision conversion
  cvt(x, x_bf, (size_t)BS_ * HID_);
  cvt(wqdown, wqd_bf, (size_t)QC_ * HID_);
  cvt(wqnope, wqn_bf, (size_t)NH_ * NOPE_ * QC_);
  cvt(wqrope, wqr_bf, (size_t)NH_ * ROPE_ * QC_);
  cvt(wkvdown, wkvd_bf, (size_t)KVC_ * HID_);
  cvt(wknope, wkn_bf, (size_t)NH_ * NOPE_ * KVC_);
  cvt(wvup, wvu_bf, (size_t)NH_ * VD_ * KVC_);
  cvt(wkrope, wkr_bf, (size_t)ROPE_ * HID_);
  cvt(wo, wo_bf, (size_t)HID_ * NH_ * VD_);

  // 1) query path
  gemm(x_bf, wqd_bf, cq_f, BS_, QC_, HID_);
  rmsnorm_rows<<<BS_, 256, 0, stream>>>(cq_f, qnormw, cq_bf, QC_);

  // 2) kv latent path
  gemm(x_bf, wkvd_bf, ckv_f, BS_, KVC_, HID_);
  rmsnorm_rows<<<BS_, 256, 0, stream>>>(ckv_f, kvnormw, ckv_bf, KVC_);
  gemm(x_bf, wkr_bf, krope_f, BS_, ROPE_, HID_);

  // 3) up-projections + RoPE + layout builds
  gemm(cq_bf, wqn_bf, tmpA, BS_, NH_ * NOPE_, QC_);
  gemm(cq_bf, wqr_bf, tmpB, BS_, NH_ * ROPE_, QC_);
  build_q<<<BS_, 256, 0, stream>>>(tmpA, tmpB, qcat);

  gemm(ckv_bf, wkn_bf, tmpA, BS_, NH_ * NOPE_, KVC_);
  build_k<<<BS_, 256, 0, stream>>>(tmpA, krope_f, kcat);

  gemm(ckv_bf, wvu_bf, tmpA, BS_, NH_ * VD_, KVC_);
  build_vt<<<4096, 256, 0, stream>>>(tmpA, vt);

  // 4) flash attention
  mla_attn<<<B_ * NH_ * (S_ / 128), 256, 0, stream>>>(qcat, kcat, vt, attn_bf);

  // 5) output projection -> f32 d_out
  gemm(attn_bf, wo_bf, out, BS_, HID_, NH_ * VD_);
}
